// TransformerNet_54228257079643
// MI455X (gfx1250) — compile-verified, matrix operand order checked by reading, not run on hardware
//
#include <hip/hip_runtime.h>
#include <hip/hip_bf16.h>

typedef __attribute__((ext_vector_type(16))) _Float16 v16h;
typedef __attribute__((ext_vector_type(8)))  float    v8f;

#define N_NODES   100000
#define N_EDGES   1000000
#define D_H       64
#define NUM_LAYERS 3
#define NUM_GRAPHS 128
#define D_OUT     10

// ---------- order-preserving float<->uint for atomic max over signed floats ----------
__device__ __forceinline__ unsigned ford(float f) {
    unsigned b = __float_as_uint(f);
    return (b & 0x80000000u) ? ~b : (b | 0x80000000u);
}
__device__ __forceinline__ float forddec(unsigned u) {
    unsigned b = (u & 0x80000000u) ? (u ^ 0x80000000u) : ~u;
    return __uint_as_float(b);
}

// ---------- fp32 -> fp16 conversion (h carried in f16 for WMMA A-operand) ----------
__global__ void k_to_f16(const float* __restrict__ x, _Float16* __restrict__ o, int n) {
    int i = blockIdx.x * blockDim.x + threadIdx.x;
    if (i < n) o[i] = (_Float16)x[i];
}

// ---------- per-layer weight prep: Wcat^T[256][64] in f16 (+ bias concat) ----------
__global__ void k_prep_w(const float* __restrict__ Wq, const float* __restrict__ Wk,
                         const float* __restrict__ Wv, const float* __restrict__ Ws,
                         const float* __restrict__ bq, const float* __restrict__ bk,
                         const float* __restrict__ bv, const float* __restrict__ bs,
                         int l, _Float16* __restrict__ WT, float* __restrict__ bcat) {
    // WT[c][i] = W_sel[l][i][c&63]  (column c of the concatenated B matrix, contiguous in K)
    for (int t = threadIdx.x; t < 256 * 64; t += blockDim.x) {
        int c = t >> 6, i = t & 63;
        int sel = c >> 6, j = c & 63;
        const float* W = (sel == 0) ? Wq : (sel == 1) ? Wk : (sel == 2) ? Wv : Ws;
        WT[t] = (_Float16)W[(size_t)l * 64 * 64 + i * 64 + j];
    }
    for (int c = threadIdx.x; c < 256; c += blockDim.x) {
        int sel = c >> 6, j = c & 63;
        const float* B = (sel == 0) ? bq : (sel == 1) ? bk : (sel == 2) ? bv : bs;
        bcat[c] = B[l * 64 + j];
    }
}

// ---------- fused Q|K|V|S GEMM: [N,64](f16) x [64,256](f16) -> 4x [N,64](f32) ----------
// one wave32 per 16x16 output tile, K=64 via two v_wmma_f32_16x16x32_f16.
// A 16x32 f16 k order used here (contiguous, not the ISA table order): lane = M (0..15), lane>>4 selects K half; 16 contiguous halves/lane.
// B 32x16 f16 k order used here: lane = N column, same K grouping -> contiguous rows of Wcat^T.
// D 16x16 f32 layout: VGPR r -> row r + 8*(lane>>4), col = lane&15.
__global__ void k_gemm_qkvs(const _Float16* __restrict__ Hf16,
                            const _Float16* __restrict__ WT,   // [256][64], contiguous K
                            const float*    __restrict__ bcat, // [256]
                            float* __restrict__ Q, float* __restrict__ K,
                            float* __restrict__ V, float* __restrict__ S) {
    int wave = (blockIdx.x * blockDim.x + threadIdx.x) >> 5;   // global wave id == tile id
    int lane = threadIdx.x & 31;
    int rowTile = wave >> 4;           // 16 column tiles cover 256 out cols
    int colTile = wave & 15;
    if (rowTile >= (N_NODES / 16)) return;   // wave-uniform guard (EXEC stays all-ones)

    int m  = lane & 15;                // A: row within tile / B: column within tile
    int kg = lane >> 4;                // K half-group (0: K 0..15, 1: K 16..31 of each step)
    int arow = rowTile * 16 + m;
    int bcol = colTile * 16 + m;

    v8f c = {};
#pragma unroll
    for (int kk = 0; kk < 64; kk += 32) {
        v16h a = *(const v16h*)(Hf16 + (size_t)arow * 64 + kk + kg * 16);
        v16h b = *(const v16h*)(WT   + (size_t)bcol * 64 + kk + kg * 16);
        c = __builtin_amdgcn_wmma_f32_16x16x32_f16(false, a, false, b, (short)0, c, false, false);
    }

    // destination selection is WAVE-UNIFORM (a 16-col tile never straddles a 64-col group);
    // assert uniformity so the pointer select is pure SALU instead of an EXEC-mask ladder.
    int sel = __builtin_amdgcn_readfirstlane(colTile >> 2);
    float* dst = (sel == 0) ? Q : (sel == 1) ? K : (sel == 2) ? V : S;
    int cc = (colTile & 3) * 16 + m;   // column within the selected 64-wide output
    float bias = bcat[bcol];
#pragma unroll
    for (int r = 0; r < 8; ++r) {
        int orow = rowTile * 16 + r + kg * 8;
        dst[(size_t)orow * 64 + cc] = c[r] + bias;
    }
}

// ---------- edge pass 1: scores + segment max (wave per edge) ----------
__global__ void k_edge_scores(const int* __restrict__ ei,
                              const float* __restrict__ Q, const float* __restrict__ K,
                              float* __restrict__ scores, unsigned* __restrict__ nodeMax) {
    int wv   = (blockIdx.x * blockDim.x + threadIdx.x) >> 5;
    int lane = threadIdx.x & 31;
    if (wv >= N_EDGES) return;
    int src = ei[wv];
    int dst = ei[N_EDGES + wv];
    float2 q = *(const float2*)(Q + (size_t)dst * 64 + lane * 2);
    float2 k = *(const float2*)(K + (size_t)src * 64 + lane * 2);
    float p = q.x * k.x + q.y * k.y;
#pragma unroll
    for (int off = 16; off; off >>= 1) p += __shfl_xor(p, off, 32);
    if (lane == 0) {
        float s = p * 0.125f;                    // 1/sqrt(64)
        scores[wv] = s;
        atomicMax(nodeMax + dst, ford(s));
    }
}

// ---------- edge pass 2: exp(score - max) + segment sum ----------
__global__ void k_edge_exp(const int* __restrict__ ei, float* __restrict__ scores,
                           const unsigned* __restrict__ nodeMax, float* __restrict__ nodeSum) {
    int e = blockIdx.x * blockDim.x + threadIdx.x;
    if (e >= N_EDGES) return;
    int dst = ei[N_EDGES + e];
    float m  = forddec(nodeMax[dst]);
    float ex = __expf(scores[e] - m);
    scores[e] = ex;
    atomicAdd(nodeSum + dst, ex);
}

// ---------- edge pass 3: agg[dst] += alpha * V[src] (wave per edge) ----------
__global__ void k_edge_scatter(const int* __restrict__ ei, const float* __restrict__ scores,
                               const float* __restrict__ nodeSum, const float* __restrict__ V,
                               float* __restrict__ agg) {
    int wv   = (blockIdx.x * blockDim.x + threadIdx.x) >> 5;
    int lane = threadIdx.x & 31;
    if (wv >= N_EDGES) return;
    int src = ei[wv];
    int dst = ei[N_EDGES + wv];
    float alpha = scores[wv] / (nodeSum[dst] + 1e-16f);
    float2 v = *(const float2*)(V + (size_t)src * 64 + lane * 2);
    atomicAdd(agg + (size_t)dst * 64 + lane * 2,     v.x * alpha);
    atomicAdd(agg + (size_t)dst * 64 + lane * 2 + 1, v.y * alpha);
}

// ---------- node update: h = relu(agg + S); write f16 for next layer; pool ----------
__global__ void k_node_update(const float* __restrict__ agg, const float* __restrict__ S,
                              const int* __restrict__ batch, _Float16* __restrict__ Hf16,
                              float* __restrict__ pooled /* [128][64] for this layer */) {
    int idx = blockIdx.x * blockDim.x + threadIdx.x;   // N*64
    if (idx >= N_NODES * 64) return;
    int n = idx >> 6, d = idx & 63;
    float h = agg[idx] + S[idx];
    h = h > 0.f ? h : 0.f;
    Hf16[idx] = (_Float16)h;
    atomicAdd(pooled + batch[n] * 64 + d, h);
}

// ---------- tiny MLP head on pooled[-1]: [128,64]->relu->[128,10] ----------
__global__ void k_head(const float* __restrict__ pooled2,
                       const float* __restrict__ W1, const float* __restrict__ b1,
                       const float* __restrict__ W2, const float* __restrict__ b2,
                       float* __restrict__ out) {
    int g = threadIdx.x;
    if (g >= NUM_GRAPHS) return;
    float p[64];
#pragma unroll 4
    for (int j = 0; j < 64; ++j) {
        float acc = b1[j];
        for (int i = 0; i < 64; ++i) acc += pooled2[g * 64 + i] * W1[i * 64 + j];
        p[j] = acc > 0.f ? acc : 0.f;
    }
    for (int o = 0; o < D_OUT; ++o) {
        float acc = b2[o];
        for (int j = 0; j < 64; ++j) acc += p[j] * W2[j * D_OUT + o];
        out[g * D_OUT + o] = acc;
    }
}

extern "C" void kernel_launch(void* const* d_in, const int* in_sizes, int n_in,
                              void* d_out, int out_size, void* d_ws, size_t ws_size,
                              hipStream_t stream) {
    const float* x     = (const float*)d_in[0];
    const int*   ei    = (const int*)  d_in[1];
    const int*   batch = (const int*)  d_in[2];
    const float* Wq = (const float*)d_in[3];  const float* bq = (const float*)d_in[4];
    const float* Wk = (const float*)d_in[5];  const float* bk = (const float*)d_in[6];
    const float* Wv = (const float*)d_in[7];  const float* bv = (const float*)d_in[8];
    const float* Ws = (const float*)d_in[9];  const float* bs = (const float*)d_in[10];
    const float* W1 = (const float*)d_in[11]; const float* b1 = (const float*)d_in[12];
    const float* W2 = (const float*)d_in[13]; const float* b2 = (const float*)d_in[14];

    // workspace carve (256B aligned suballocations)
    char* ws = (char*)d_ws;
    auto carve = [&](size_t bytes) { char* p = ws; ws += (bytes + 255) & ~(size_t)255; return p; };
    _Float16* Hf16   = (_Float16*)carve((size_t)N_NODES * 64 * 2);
    float*    Q      = (float*)   carve((size_t)N_NODES * 64 * 4);
    float*    K      = (float*)   carve((size_t)N_NODES * 64 * 4);
    float*    V      = (float*)   carve((size_t)N_NODES * 64 * 4);
    float*    S      = (float*)   carve((size_t)N_NODES * 64 * 4);
    float*    agg    = (float*)   carve((size_t)N_NODES * 64 * 4);
    float*    scores = (float*)   carve((size_t)N_EDGES * 4);
    unsigned* nodeMax= (unsigned*)carve((size_t)N_NODES * 4);
    float*    nodeSum= (float*)   carve((size_t)N_NODES * 4);
    _Float16* WT     = (_Float16*)carve(256 * 64 * 2);
    float*    bcat   = (float*)   carve(256 * 4);

    float* out    = (float*)d_out;            // [128,10]
    float* pooled = out + NUM_GRAPHS * D_OUT; // [3][128][64]

    hipMemsetAsync(d_out, 0, (size_t)out_size * sizeof(float), stream);

    k_to_f16<<<(N_NODES * 64 + 255) / 256, 256, 0, stream>>>(x, Hf16, N_NODES * 64);

    const int gemmBlocks  = (N_NODES / 16) * 16 / 8;   // 100000 tiles / 8 waves per block
    const int edgeWaveBlk = N_EDGES / 8;               // wave-per-edge, 8 waves per block

    for (int l = 0; l < NUM_LAYERS; ++l) {
        k_prep_w<<<1, 256, 0, stream>>>(Wq, Wk, Wv, Ws, bq, bk, bv, bs, l, WT, bcat);
        k_gemm_qkvs<<<gemmBlocks, 256, 0, stream>>>(Hf16, WT, bcat, Q, K, V, S);

        hipMemsetAsync(nodeMax, 0, (size_t)N_NODES * 4, stream); // 0 == below ord(-inf): identity
        hipMemsetAsync(nodeSum, 0, (size_t)N_NODES * 4, stream);
        hipMemsetAsync(agg,     0, (size_t)N_NODES * 64 * 4, stream);

        k_edge_scores <<<edgeWaveBlk, 256, 0, stream>>>(ei, Q, K, scores, nodeMax);
        k_edge_exp    <<<(N_EDGES + 255) / 256, 256, 0, stream>>>(ei, scores, nodeMax, nodeSum);
        k_edge_scatter<<<edgeWaveBlk, 256, 0, stream>>>(ei, scores, nodeSum, V, agg);

        k_node_update<<<(N_NODES * 64 + 255) / 256, 256, 0, stream>>>(
            agg, S, batch, Hf16, pooled + (size_t)l * NUM_GRAPHS * 64);
    }

    k_head<<<1, 128, 0, stream>>>(pooled + 2 * NUM_GRAPHS * 64, W1, b1, W2, b2, out);
}
